// VisionTransformer_46050639347819
// MI455X (gfx1250) — compile-verified, matrix operand order checked by reading, not run on hardware
//
#include <hip/hip_runtime.h>

// ---------------------------------------------------------------------------
// ViT attention block for MI455X (gfx1250, wave32, WMMA, async-to-LDS).
// Pipeline:
//   0) convert W{q,k,v,o} fp32 -> transposed bf16 (WT[c][k])
//   1) LayerNorm q -> qn(bf16);  LayerNorm kv -> kn,vn (bf16, stats shared)
//   2) WMMA GEMMs (double-buffered async-to-LDS): Q,K -> [bh][n][64] bf16,
//      V -> transposed [bh][64][n] bf16
//   3) histogram of scaled logits per head (WMMA recompute, no NxN in HBM)
//   4) kth-value threshold per head from histogram
//   5) flash attention with threshold mask (WMMA S, WMMA P*V, online softmax)
//   6) WMMA out-projection + bias + residual (fp32) -> final LayerNorm -> d_out
// ---------------------------------------------------------------------------

typedef __attribute__((ext_vector_type(16))) __bf16 v16bf;
typedef __attribute__((ext_vector_type(8)))  float  v8f;
typedef __attribute__((ext_vector_type(4)))  int    vi4;

#define DIM_C   768
#define MROWS   8192          // B*N
#define NTOK    1024
#define NHEADS  12
#define HD      64
#define BHN     96            // B*H
#define SCALE   0.125f        // 1/sqrt(64)
#define HIST_NB 1024
#define HIST_LO (-16.0f)
#define HIST_BW 0.03125f
#define HIST_IBW 32.0f
#define KTH_RANK 104858       // numel - int(0.9*numel), 1-indexed kth smallest

static __device__ __forceinline__ unsigned short f2bf(float f) {
    union { float f; unsigned u; } v; v.f = f;
    unsigned r = v.u + 0x7fffu + ((v.u >> 16) & 1u);
    return (unsigned short)(r >> 16);
}

static __device__ __forceinline__ v8f zero8() {
    v8f z;
    #pragma unroll
    for (int i = 0; i < 8; ++i) z[i] = 0.0f;
    return z;
}

static __device__ __forceinline__ v8f wmma_bf16(v16bf a, v16bf b, v8f c) {
    return __builtin_amdgcn_wmma_f32_16x16x32_bf16(false, a, false, b, (short)0, c,
                                                   false, false);
}

// ---- CDNA5 async global->LDS (tracked by ASYNCcnt) -------------------------
// LDS operand: low 32 bits of the generic pointer to __shared__ data are the
// wave-relative LDS byte offset (ISA 10.2 aperture mapping).
static __device__ __forceinline__ void async_ld_b128(const void* gptr,
                                                     const void* lptr) {
    unsigned lds = (unsigned)(unsigned long long)lptr;
    unsigned long long ga = (unsigned long long)gptr;
    asm volatile("global_load_async_to_lds_b128 %0, %1, off"
                 :: "v"(lds), "v"(ga) : "memory");
}
static __device__ __forceinline__ void wait_async0() {
    asm volatile("s_wait_asynccnt 0x0" ::: "memory");
}
static __device__ __forceinline__ void wait_async3() {
    asm volatile("s_wait_asynccnt 0x3" ::: "memory");
}

// Load a 16x32 bf16 A/B fragment from a row-major LDS tile (stride in halves).
// Lanes 0-15: row = base+lane, k halves {0..7,16..23}; lanes 16-31: {8..15,24..31}.
static __device__ __forceinline__ v16bf frag16(const unsigned short* lds,
                                               int rowBase, int stride, int kOff) {
    int lane = threadIdx.x & 31;
    const unsigned short* p =
        lds + (rowBase + (lane & 15)) * stride + kOff + ((lane & 16) ? 8 : 0);
    union { vi4 q[2]; v16bf v; } u;
    u.q[0] = *(const vi4*)(p);
    u.q[1] = *(const vi4*)(p + 16);
    return u.v;
}

// ------------------------- prep: weight convert/transpose -------------------
__global__ __launch_bounds__(256) void wcvt_kernel(const float* __restrict__ W,
                                                   unsigned short* __restrict__ WT) {
    int idx = blockIdx.x * 256 + threadIdx.x;          // 768*768 elements
    int k = idx / DIM_C, c = idx % DIM_C;
    WT[(size_t)c * DIM_C + k] = f2bf(W[idx]);
}

// ------------------------- LayerNorm -> bf16 (1 or 2 outputs) ---------------
__global__ __launch_bounds__(256)
void ln_bf16_kernel(const float* __restrict__ x,
                    const float* __restrict__ w1, const float* __restrict__ b1,
                    unsigned short* __restrict__ y1,
                    const float* __restrict__ w2, const float* __restrict__ b2,
                    unsigned short* __restrict__ y2) {
    __shared__ float red[8], red2[8];
    int row = blockIdx.x, t = threadIdx.x;
    const float* xr = x + (size_t)row * DIM_C;
    float a0 = xr[t], a1 = xr[t + 256], a2 = xr[t + 512];
    float s = a0 + a1 + a2;
    #pragma unroll
    for (int o = 16; o; o >>= 1) s += __shfl_xor(s, o, 32);
    if ((t & 31) == 0) red[t >> 5] = s;
    __syncthreads();
    float tot = 0.f;
    #pragma unroll
    for (int i = 0; i < 8; ++i) tot += red[i];
    float mu = tot * (1.0f / DIM_C);
    float d0 = a0 - mu, d1 = a1 - mu, d2 = a2 - mu;
    float ss = d0 * d0 + d1 * d1 + d2 * d2;
    #pragma unroll
    for (int o = 16; o; o >>= 1) ss += __shfl_xor(ss, o, 32);
    if ((t & 31) == 0) red2[t >> 5] = ss;
    __syncthreads();
    float var = 0.f;
    #pragma unroll
    for (int i = 0; i < 8; ++i) var += red2[i];
    float rstd = rsqrtf(var * (1.0f / DIM_C) + 1e-5f);
    size_t base = (size_t)row * DIM_C;
    y1[base + t      ] = f2bf(d0 * rstd * w1[t      ] + b1[t      ]);
    y1[base + t + 256] = f2bf(d1 * rstd * w1[t + 256] + b1[t + 256]);
    y1[base + t + 512] = f2bf(d2 * rstd * w1[t + 512] + b1[t + 512]);
    if (y2) {
        y2[base + t      ] = f2bf(d0 * rstd * w2[t      ] + b2[t      ]);
        y2[base + t + 256] = f2bf(d1 * rstd * w2[t + 256] + b2[t + 256]);
        y2[base + t + 512] = f2bf(d2 * rstd * w2[t + 512] + b2[t + 512]);
    }
}

// ------------------------- final LayerNorm -> fp32 --------------------------
__global__ __launch_bounds__(256)
void ln_final_kernel(const float* __restrict__ x, const float* __restrict__ w,
                     const float* __restrict__ b, float* __restrict__ y) {
    __shared__ float red[8], red2[8];
    int row = blockIdx.x, t = threadIdx.x;
    const float* xr = x + (size_t)row * DIM_C;
    float a0 = xr[t], a1 = xr[t + 256], a2 = xr[t + 512];
    float s = a0 + a1 + a2;
    #pragma unroll
    for (int o = 16; o; o >>= 1) s += __shfl_xor(s, o, 32);
    if ((t & 31) == 0) red[t >> 5] = s;
    __syncthreads();
    float tot = 0.f;
    #pragma unroll
    for (int i = 0; i < 8; ++i) tot += red[i];
    float mu = tot * (1.0f / DIM_C);
    float d0 = a0 - mu, d1 = a1 - mu, d2 = a2 - mu;
    float ss = d0 * d0 + d1 * d1 + d2 * d2;
    #pragma unroll
    for (int o = 16; o; o >>= 1) ss += __shfl_xor(ss, o, 32);
    if ((t & 31) == 0) red2[t >> 5] = ss;
    __syncthreads();
    float var = 0.f;
    #pragma unroll
    for (int i = 0; i < 8; ++i) var += red2[i];
    float rstd = rsqrtf(var * (1.0f / DIM_C) + 1e-5f);
    size_t base = (size_t)row * DIM_C;
    y[base + t      ] = d0 * rstd * w[t      ] + b[t      ];
    y[base + t + 256] = d1 * rstd * w[t + 256] + b[t + 256];
    y[base + t + 512] = d2 * rstd * w[t + 512] + b[t + 512];
}

// ------------------------- WMMA GEMM: Y = X(bf16) @ WT^T + bias -------------
// Double-buffered async-to-LDS pipeline over the K dimension.
// MODE 0: out bf16, head layout [bh][n][64]   (Q, K)
// MODE 1: out bf16, transposed  [bh][64][n]   (V)
// MODE 2: out fp32, row layout, + residual    (out-projection)
template <int MODE>
__global__ __launch_bounds__(256)
void gemm_wmma(const unsigned short* __restrict__ X,
               const unsigned short* __restrict__ WT,
               const float* __restrict__ bias, void* __restrict__ outp,
               const float* __restrict__ residual) {
    __shared__ unsigned short As[2][128 * 40];
    __shared__ unsigned short Bs[2][64 * 40];
    const int tid = threadIdx.x;
    const int wave = tid >> 5, wm = wave & 3, wn = wave >> 2;
    const int row0 = blockIdx.y * 128;
    const int col0 = blockIdx.x * 64;

    v8f acc[2][2];
    #pragma unroll
    for (int i = 0; i < 2; ++i)
        #pragma unroll
        for (int j = 0; j < 2; ++j) acc[i][j] = zero8();

    // 3 async-to-LDS b128 instructions per wave per tile
    auto issueTile = [&](int k0, int buf) {
        #pragma unroll
        for (int c = 0; c < 2; ++c) {               // A: 128x32 halves
            int ch = tid + c * 256;
            int r = ch >> 2, cc = (ch & 3) * 8;
            async_ld_b128(&X[(size_t)(row0 + r) * DIM_C + k0 + cc],
                          &As[buf][r * 40 + cc]);
        }
        int cB = tid >> 2, kk = (tid & 3) * 8;      // B: 64x32 halves (WT rows)
        async_ld_b128(&WT[(size_t)(col0 + cB) * DIM_C + k0 + kk],
                      &Bs[buf][cB * 40 + kk]);
    };

    issueTile(0, 0);
    const int NT = DIM_C / 32;                      // 24
    for (int kt = 0; kt < NT; ++kt) {
        const int cur = kt & 1;
        if (kt + 1 < NT) {
            issueTile((kt + 1) * 32, 1 - cur);      // prefetch next tile
            wait_async3();                          // current tile resident
        } else {
            wait_async0();
        }
        __syncthreads();
        #pragma unroll
        for (int i = 0; i < 2; ++i) {
            v16bf a = frag16(As[cur], wm * 32 + i * 16, 40, 0);
            #pragma unroll
            for (int j = 0; j < 2; ++j) {
                v16bf b = frag16(Bs[cur], wn * 32 + j * 16, 40, 0);
                acc[i][j] = wmma_bf16(a, b, acc[i][j]);
            }
        }
        __syncthreads();                            // protect WAR on buf 1-cur
    }

    const int lane = tid & 31;
    const int nOff = lane & 15;
    const int mAdd = (lane & 16) ? 8 : 0;
    #pragma unroll
    for (int i = 0; i < 2; ++i)
        #pragma unroll
        for (int j = 0; j < 2; ++j)
            #pragma unroll
            for (int e = 0; e < 8; ++e) {
                int r = row0 + wm * 32 + i * 16 + e + mAdd;
                int c = col0 + wn * 32 + j * 16 + nOff;
                float v = acc[i][j][e] + bias[c];
                if (MODE == 2) {
                    size_t a = (size_t)r * DIM_C + c;
                    ((float*)outp)[a] = v + residual[a];
                } else {
                    int b = r >> 10, n = r & 1023;
                    int h = c >> 6, d = c & 63;
                    size_t base = (size_t)(b * NHEADS + h) << 16;  // *65536
                    size_t a = (MODE == 0) ? base + (size_t)n * HD + d
                                           : base + (size_t)d * NTOK + n;
                    ((unsigned short*)outp)[a] = f2bf(v);
                }
            }
}

// ------------------------- logits histogram (per head) ----------------------
__global__ __launch_bounds__(256)
void hist_kernel(const unsigned short* __restrict__ Qh,
                 const unsigned short* __restrict__ Kh,
                 unsigned int* __restrict__ hist) {
    __shared__ unsigned short Qs[128 * 72];
    __shared__ unsigned short Ks[64 * 72];
    __shared__ unsigned int lh[HIST_NB];
    const int tid = threadIdx.x;
    const int bh = blockIdx.z, q0 = blockIdx.y * 128, kk0 = blockIdx.x * 64;
    const unsigned short* Qb = Qh + ((size_t)bh << 16);
    const unsigned short* Kb = Kh + ((size_t)bh << 16);
    #pragma unroll
    for (int i = 0; i < 4; ++i) lh[tid + i * 256] = 0u;
    #pragma unroll
    for (int c = 0; c < 4; ++c) {
        int ch = tid + c * 256, r = ch >> 3, cc = (ch & 7) * 8;
        async_ld_b128(&Qb[(size_t)(q0 + r) * HD + cc], &Qs[r * 72 + cc]);
    }
    #pragma unroll
    for (int c = 0; c < 2; ++c) {
        int ch = tid + c * 256, r = ch >> 3, cc = (ch & 7) * 8;
        async_ld_b128(&Kb[(size_t)(kk0 + r) * HD + cc], &Ks[r * 72 + cc]);
    }
    wait_async0();
    __syncthreads();
    const int wave = tid >> 5, wm = wave & 3, wn = wave >> 2;
    #pragma unroll
    for (int i = 0; i < 2; ++i)
        #pragma unroll
        for (int j = 0; j < 2; ++j) {
            v8f s = zero8();
            #pragma unroll
            for (int d0 = 0; d0 < 64; d0 += 32) {
                v16bf a = frag16(Qs, wm * 32 + i * 16, 72, d0);
                v16bf b = frag16(Ks, wn * 32 + j * 16, 72, d0);
                s = wmma_bf16(a, b, s);
            }
            #pragma unroll
            for (int e = 0; e < 8; ++e) {
                float v = s[e] * SCALE;
                int bin = (int)((v - HIST_LO) * HIST_IBW);
                bin = bin < 0 ? 0 : (bin > HIST_NB - 1 ? HIST_NB - 1 : bin);
                atomicAdd(&lh[bin], 1u);
            }
        }
    __syncthreads();
    unsigned int* gh = hist + (size_t)bh * HIST_NB;
    #pragma unroll
    for (int i = 0; i < 4; ++i) {
        unsigned v = lh[tid + i * 256];
        if (v) atomicAdd(&gh[tid + i * 256], v);
    }
}

__global__ void zerou32_kernel(unsigned int* p, int n) {
    int i = blockIdx.x * 256 + threadIdx.x;
    if (i < n) p[i] = 0u;
}

__global__ void thr_kernel(const unsigned int* __restrict__ hist,
                           float* __restrict__ thr) {
    int h = threadIdx.x;
    if (h >= BHN) return;
    const unsigned int* H = hist + (size_t)h * HIST_NB;
    unsigned cum = 0; int bin = HIST_NB - 1;
    for (int i = 0; i < HIST_NB; ++i) {
        cum += H[i];
        if (cum >= (unsigned)KTH_RANK) { bin = i; break; }
    }
    thr[h] = HIST_LO + bin * HIST_BW;   // lower edge of bin holding kth value
}

// ------------------------- flash attention with threshold -------------------
__global__ __launch_bounds__(256)
void flash_kernel(const unsigned short* __restrict__ Qh,
                  const unsigned short* __restrict__ Kh,
                  const unsigned short* __restrict__ VTh,
                  const float* __restrict__ thrArr,
                  unsigned short* __restrict__ attn_out) {
    __shared__ unsigned short Qs[128 * 72];
    __shared__ unsigned short Ks[64 * 72];
    __shared__ unsigned short Ps[128 * 72];
    __shared__ unsigned short VTs[64 * 72];
    __shared__ float Ss[128 * 66];
    __shared__ float mArr[128], lArr[128], scArr[128];

    const int tid = threadIdx.x, wave = tid >> 5, wm = wave & 3, wn = wave >> 2;
    const int lane = tid & 31, nOff = lane & 15, mAdd = (lane & 16) ? 8 : 0;
    const int bh = blockIdx.y, q0 = blockIdx.x * 128;
    const float thr = thrArr[bh];
    const unsigned short* Qb = Qh + ((size_t)bh << 16);
    const unsigned short* Kb = Kh + ((size_t)bh << 16);
    const unsigned short* Vb = VTh + ((size_t)bh << 16);

    #pragma unroll
    for (int c = 0; c < 4; ++c) {
        int ch = tid + c * 256, r = ch >> 3, cc = (ch & 7) * 8;
        async_ld_b128(&Qb[(size_t)(q0 + r) * HD + cc], &Qs[r * 72 + cc]);
    }
    if (tid < 128) { mArr[tid] = -INFINITY; lArr[tid] = 0.0f; }
    v8f O[2][2];
    #pragma unroll
    for (int i = 0; i < 2; ++i)
        #pragma unroll
        for (int j = 0; j < 2; ++j) O[i][j] = zero8();
    wait_async0();
    __syncthreads();

    for (int kt = 0; kt < 16; ++kt) {
        const int kk0 = kt * 64;
        #pragma unroll
        for (int c = 0; c < 2; ++c) {          // K tile [kk][d]
            int ch = tid + c * 256, r = ch >> 3, cc = (ch & 7) * 8;
            async_ld_b128(&Kb[(size_t)(kk0 + r) * HD + cc], &Ks[r * 72 + cc]);
        }
        #pragma unroll
        for (int c = 0; c < 2; ++c) {          // V^T tile [d][kk]
            int ch = tid + c * 256, r = ch >> 3, cc = (ch & 7) * 8;
            async_ld_b128(&Vb[(size_t)r * NTOK + kk0 + cc], &VTs[r * 72 + cc]);
        }
        wait_async0();
        __syncthreads();

        // S = Q K^T (scaled), staged to LDS
        #pragma unroll
        for (int i = 0; i < 2; ++i)
            #pragma unroll
            for (int j = 0; j < 2; ++j) {
                v8f s = zero8();
                #pragma unroll
                for (int d0 = 0; d0 < 64; d0 += 32) {
                    v16bf a = frag16(Qs, wm * 32 + i * 16, 72, d0);
                    v16bf b = frag16(Ks, wn * 32 + j * 16, 72, d0);
                    s = wmma_bf16(a, b, s);
                }
                #pragma unroll
                for (int e = 0; e < 8; ++e)
                    Ss[(wm * 32 + i * 16 + e + mAdd) * 66 +
                       wn * 32 + j * 16 + nOff] = s[e] * SCALE;
            }
        __syncthreads();

        // online softmax per query row: mask -> clip -> max -> exp -> P(bf16)
        if (tid < 128) {
            float* Sr = &Ss[tid * 66];
            float mOld = mArr[tid], mNew = mOld;
            for (int c = 0; c < 64; ++c) {
                float v = Sr[c];
                v = (v < thr) ? -10000.0f : v;
                v = fminf(fmaxf(v, -10000.0f), 10000.0f);
                Sr[c] = v;
                mNew = fmaxf(mNew, v);
            }
            float sc = expf(mOld - mNew);     // -inf -> 0 on first tile
            float l = lArr[tid] * sc;
            for (int c = 0; c < 64; ++c) {
                float p = expf(Sr[c] - mNew);
                l += p;
                Ps[tid * 72 + c] = f2bf(p);
            }
            mArr[tid] = mNew; lArr[tid] = l; scArr[tid] = sc;
        }
        __syncthreads();

        // rescale O and accumulate O += P @ V
        #pragma unroll
        for (int i = 0; i < 2; ++i)
            #pragma unroll
            for (int j = 0; j < 2; ++j)
                #pragma unroll
                for (int e = 0; e < 8; ++e)
                    O[i][j][e] *= scArr[wm * 32 + i * 16 + e + mAdd];
        #pragma unroll
        for (int i = 0; i < 2; ++i)
            #pragma unroll
            for (int j = 0; j < 2; ++j) {
                #pragma unroll
                for (int k0 = 0; k0 < 64; k0 += 32) {
                    v16bf a = frag16(Ps, wm * 32 + i * 16, 72, k0);
                    v16bf b = frag16(VTs, wn * 32 + j * 16, 72, k0);
                    O[i][j] = wmma_bf16(a, b, O[i][j]);
                }
            }
        __syncthreads();                      // protects Ks/VTs/Ps WAR next tile
    }

    // epilogue: normalize by l, write bf16 [b][n][h*64+d]
    const int bb = bh / NHEADS, hh = bh % NHEADS;
    #pragma unroll
    for (int i = 0; i < 2; ++i)
        #pragma unroll
        for (int j = 0; j < 2; ++j)
            #pragma unroll
            for (int e = 0; e < 8; ++e) {
                int r = wm * 32 + i * 16 + e + mAdd;
                int n = q0 + r;
                int d = wn * 32 + j * 16 + nOff;
                float v = O[i][j][e] / lArr[r];
                attn_out[(size_t)(bb * NTOK + n) * DIM_C + hh * HD + d] = f2bf(v);
            }
}

// ---------------------------------------------------------------------------
extern "C" void kernel_launch(void* const* d_in, const int* in_sizes, int n_in,
                              void* d_out, int out_size, void* d_ws, size_t ws_size,
                              hipStream_t stream) {
    const float* q_in  = (const float*)d_in[0];
    const float* kv_in = (const float*)d_in[1];
    const float* ln_q_w = (const float*)d_in[2];
    const float* ln_q_b = (const float*)d_in[3];
    const float* ln_k_w = (const float*)d_in[4];
    const float* ln_k_b = (const float*)d_in[5];
    const float* ln_v_w = (const float*)d_in[6];
    const float* ln_v_b = (const float*)d_in[7];
    const float* ln_o_w = (const float*)d_in[8];
    const float* ln_o_b = (const float*)d_in[9];
    const float* Wq = (const float*)d_in[10];
    const float* bq = (const float*)d_in[11];
    const float* Wk = (const float*)d_in[12];
    const float* bk = (const float*)d_in[13];
    const float* Wv = (const float*)d_in[14];
    const float* bv = (const float*)d_in[15];
    const float* Wo = (const float*)d_in[16];
    const float* bo = (const float*)d_in[17];

    // workspace layout (all 256B aligned); total ~113 MB
    char* ws = (char*)d_ws;
    auto take = [&](size_t bytes) {
        char* p = ws;
        ws += (bytes + 255) & ~(size_t)255;
        return p;
    };
    const size_t WBYTES = (size_t)DIM_C * DIM_C * 2;       // bf16 768x768
    const size_t XBYTES = (size_t)MROWS * DIM_C * 2;       // bf16 8192x768
    unsigned short* WTq = (unsigned short*)take(WBYTES);
    unsigned short* WTk = (unsigned short*)take(WBYTES);
    unsigned short* WTv = (unsigned short*)take(WBYTES);
    unsigned short* WTo = (unsigned short*)take(WBYTES);
    unsigned short* qn  = (unsigned short*)take(XBYTES);
    unsigned short* kn  = (unsigned short*)take(XBYTES);
    unsigned short* vn  = (unsigned short*)take(XBYTES);
    unsigned short* Qh  = (unsigned short*)take(XBYTES);   // [bh][n][64]
    unsigned short* Kh  = (unsigned short*)take(XBYTES);   // [bh][n][64]
    unsigned short* VTh = (unsigned short*)take(XBYTES);   // [bh][64][n]
    unsigned short* att = (unsigned short*)take(XBYTES);   // [b][n][768]
    unsigned int*   hist = (unsigned int*)take((size_t)BHN * HIST_NB * 4);
    float*          thr  = (float*)take(BHN * 4);
    float*          tmp  = (float*)take((size_t)MROWS * DIM_C * 4);

    const dim3 blk(256);

    // 0) weight convert+transpose
    wcvt_kernel<<<2304, blk, 0, stream>>>(Wq, WTq);
    wcvt_kernel<<<2304, blk, 0, stream>>>(Wk, WTk);
    wcvt_kernel<<<2304, blk, 0, stream>>>(Wv, WTv);
    wcvt_kernel<<<2304, blk, 0, stream>>>(Wo, WTo);

    // 1) LayerNorms
    ln_bf16_kernel<<<MROWS, blk, 0, stream>>>(q_in, ln_q_w, ln_q_b, qn,
                                              nullptr, nullptr, nullptr);
    ln_bf16_kernel<<<MROWS, blk, 0, stream>>>(kv_in, ln_k_w, ln_k_b, kn,
                                              ln_v_w, ln_v_b, vn);

    // 2) QKV projections (WMMA, async double-buffered)
    const dim3 ggrid(DIM_C / 64, MROWS / 128);
    gemm_wmma<0><<<ggrid, blk, 0, stream>>>(qn, WTq, bq, Qh, nullptr);
    gemm_wmma<0><<<ggrid, blk, 0, stream>>>(kn, WTk, bk, Kh, nullptr);
    gemm_wmma<1><<<ggrid, blk, 0, stream>>>(vn, WTv, bv, VTh, nullptr);

    // 3) logits histogram per head (WMMA recompute, no NxN materialization)
    zerou32_kernel<<<(BHN * HIST_NB + 255) / 256, blk, 0, stream>>>(
        hist, BHN * HIST_NB);
    hist_kernel<<<dim3(NTOK / 64, NTOK / 128, BHN), blk, 0, stream>>>(Qh, Kh, hist);

    // 4) per-head kth-value threshold
    thr_kernel<<<1, 128, 0, stream>>>(hist, thr);

    // 5) flash attention with sparsity threshold
    flash_kernel<<<dim3(NTOK / 128, BHN), blk, 0, stream>>>(Qh, Kh, VTh, thr, att);

    // 6) out projection + bias + residual, then final LayerNorm
    gemm_wmma<2><<<ggrid, blk, 0, stream>>>(att, WTo, bo, tmp, q_in);
    ln_final_kernel<<<MROWS, blk, 0, stream>>>(tmp, ln_o_w, ln_o_b, (float*)d_out);
}
